// Decoder_49735721288430
// MI455X (gfx1250) — compile-verified, matrix operand order checked by reading, not run on hardware
//
#include <hip/hip_runtime.h>
#include <math.h>

#define BATCH   512
#define TSTEPS  200
#define HID     512
#define LATD    128
#define XHLD    544            // 512 h + 7 prev + 25 zero pad = 17 K-tiles of 32
#define G4      2048
#define JDIM    7
#define ODIM    3
#define FKOUT   24
#define COMB    27
#define JOINT_BASE (BATCH*TSTEPS*COMB)

typedef __attribute__((ext_vector_type(16))) __bf16 v16bf;
typedef __attribute__((ext_vector_type(8)))  __bf16 v8bf;
typedef __attribute__((ext_vector_type(8)))  float  v8f;

__device__ __forceinline__ v8f vzero8() {
  v8f v;
#pragma unroll
  for (int i = 0; i < 8; ++i) v[i] = 0.f;
  return v;
}

// round-to-nearest-even f32 -> bf16 bit pattern
__device__ __forceinline__ unsigned short f2bf(float f) {
  unsigned int u = __float_as_uint(f);
  u += 0x7fffu + ((u >> 16) & 1u);
  return (unsigned short)(u >> 16);
}

__device__ __forceinline__ float sigm(float x) { return 1.f / (1.f + __expf(-x)); }

// Load one 16(rows)x32(K) bf16 tile from a row-major [rows][ld] buffer.
// CDNA5 16-bit A/B layout: lanes 0-15 hold row (lane&15), K = 0..7 then 16..23;
// lanes 16-31 hold the same rows, K = 8..15 then 24..31; 2 bf16 packed per VGPR.
__device__ __forceinline__ v16bf load_tile16x32(const unsigned short* base,
                                                int ld, int row0, int col0, int lane) {
  const int r  = row0 + (lane & 15);
  const int kh = (lane >> 4) & 1;
  const unsigned short* q = base + (size_t)r * ld + col0 + kh * 8;
  v8bf lo = *(const v8bf*)(q);        // K offsets kh*8 .. kh*8+7
  v8bf hi = *(const v8bf*)(q + 16);   // K offsets 16+kh*8 .. 16+kh*8+7
  return __builtin_shufflevector(lo, hi, 0,1,2,3,4,5,6,7,8,9,10,11,12,13,14,15);
}

__device__ __forceinline__ v8f wmma_bf16(v16bf a, v16bf b, v8f c) {
  return __builtin_amdgcn_wmma_f32_16x16x32_bf16(false, a, false, b, (short)0, c, false, false);
}

// ---------------- one-time conversion / packing kernels ----------------

__global__ void f2bf_copy_kernel(const float* __restrict__ src,
                                 unsigned short* __restrict__ dst, int n) {
  int i = blockIdx.x * 256 + threadIdx.x;
  if (i < n) dst[i] = f2bf(src[i]);
}

// Wc[n][k] : k<512 -> W_hh[n][k]; 512<=k<519 -> W_ih[n][k-512]; else 0   (n in [0,2048))
__global__ void build_Wc_kernel(unsigned short* __restrict__ dst,
                                const float* __restrict__ Whh,
                                const float* __restrict__ Wih) {
  int i = blockIdx.x * 256 + threadIdx.x;
  if (i >= G4 * XHLD) return;
  int n = i / XHLD, k = i - n * XHLD;
  float v = 0.f;
  if (k < HID)              v = Whh[(size_t)n * HID + k];
  else if (k < HID + JDIM)  v = Wih[(size_t)n * (JDIM + LATD) + (k - HID)];
  dst[i] = f2bf(v);
}

// W_ihz[n][k] = W_ih[n][7+k], k in [0,128)
__global__ void build_Wihz_kernel(unsigned short* __restrict__ dst,
                                  const float* __restrict__ Wih) {
  int i = blockIdx.x * 256 + threadIdx.x;
  if (i >= G4 * LATD) return;
  int n = i >> 7, k = i & 127;
  dst[i] = f2bf(Wih[(size_t)n * (JDIM + LATD) + JDIM + k]);
}

// Wao[16][1024]: rows 0..6 = [Wa2 | 0], rows 7..9 = [0 | Wo2], rows 10..15 = 0
__global__ void build_Wao_kernel(unsigned short* __restrict__ dst,
                                 const float* __restrict__ Wa2,
                                 const float* __restrict__ Wo2) {
  int i = blockIdx.x * 256 + threadIdx.x;
  if (i >= 16 * 1024) return;
  int r = i >> 10, k = i & 1023;
  float v = 0.f;
  if (r < JDIM && k < HID)                           v = Wa2[(size_t)r * HID + k];
  else if (r >= JDIM && r < JDIM + ODIM && k >= HID) v = Wo2[(size_t)(r - JDIM) * HID + (k - HID)];
  dst[i] = f2bf(v);
}

// prev0 + zero padding for both ping-pong xh buffers (cols 512..543)
__global__ void init_prev_kernel(unsigned short* __restrict__ xh0,
                                 unsigned short* __restrict__ xh1,
                                 const float* __restrict__ st,
                                 const float* __restrict__ lo,
                                 const float* __restrict__ hi) {
  int i = blockIdx.x * 256 + threadIdx.x;
  if (i >= BATCH * 32) return;
  int b = i >> 5, j = i & 31;
  float v = 0.f;
  if (j < JDIM) {
    float rng = (hi[j] - lo[j]) * 0.5f;
    float mn  = (hi[j] + lo[j]) * 0.5f;
    v = (st[j] - mn) / rng;
  }
  xh0[(size_t)b * XHLD + HID + j] = f2bf(v);
  xh1[(size_t)b * XHLD + HID + j] = 0;   // pads; prev cols overwritten by head step 0
}

// ---------------- one-time WMMA kernels ----------------

// h0 = z@W_zh.T + b_zh (bf16 into xh0), c0 = z@W_zc.T + b_zc (f32)
__global__ void __launch_bounds__(32) init_hc_kernel(
    const unsigned short* __restrict__ zb,
    const unsigned short* __restrict__ Wzhb,
    const unsigned short* __restrict__ Wzcb,
    const float* __restrict__ bzh, const float* __restrict__ bzc,
    unsigned short* __restrict__ xh0, float* __restrict__ cbuf) {
  const int lane = threadIdx.x;
  const int m0 = blockIdx.x * 16, n0 = blockIdx.y * 16;
  v8f ah = vzero8(), ac = vzero8();
#pragma unroll
  for (int kt = 0; kt < 4; ++kt) {
    v16bf A  = load_tile16x32(zb,   LATD, m0, kt * 32, lane);
    v16bf Bh = load_tile16x32(Wzhb, LATD, n0, kt * 32, lane);
    v16bf Bc = load_tile16x32(Wzcb, LATD, n0, kt * 32, lane);
    ah = wmma_bf16(A, Bh, ah);
    ac = wmma_bf16(A, Bc, ac);
  }
  const int col = lane & 15, mb = (lane >> 4) * 8;
#pragma unroll
  for (int r = 0; r < 8; ++r) {
    const int m = m0 + mb + r, n = n0 + col;
    xh0[(size_t)m * XHLD + n]  = f2bf(ah[r] + bzh[n]);
    cbuf[(size_t)m * HID + n]  = ac[r] + bzc[n];
  }
}

// zpart[b][n] = b_ih[n] + b_hh[n] + z@W_ih[:,7:].T   (n in [0,2048))
__global__ void __launch_bounds__(32) zpart_kernel(
    const unsigned short* __restrict__ zb,
    const unsigned short* __restrict__ Wihz,
    const float* __restrict__ bih, const float* __restrict__ bhh,
    float* __restrict__ zpart) {
  const int lane = threadIdx.x;
  const int m0 = blockIdx.x * 16, n0 = blockIdx.y * 16;
  v8f acc = vzero8();
#pragma unroll
  for (int kt = 0; kt < 4; ++kt) {
    v16bf A = load_tile16x32(zb,   LATD, m0, kt * 32, lane);
    v16bf B = load_tile16x32(Wihz, LATD, n0, kt * 32, lane);
    acc = wmma_bf16(A, B, acc);
  }
  const int col = lane & 15, mb = (lane >> 4) * 8;
#pragma unroll
  for (int r = 0; r < 8; ++r) {
    const int m = m0 + mb + r, n = n0 + col;
    zpart[(size_t)m * G4 + n] = acc[r] + bih[n] + bhh[n];
  }
}

// ---------------- per-step kernels ----------------

// Fused gate GEMM + LSTM cell. One wave computes the i/f/g/o 16x16 tiles for
// TWO batch tiles (M=32) sharing the four B (weight) tiles: 8 WMMA per
// 6 tile loads, halving per-step weight traffic from L2.
// Reads xh_r = [h_{t-1} | prev_{t-1} | 0], writes h_t (bf16) into xh_w.
__global__ void __launch_bounds__(32) lstm_gates_kernel(
    const unsigned short* __restrict__ xh_r,
    unsigned short* __restrict__ xh_w,
    const unsigned short* __restrict__ Wc,     // [2048][544] = [W_hh | W_ih[:,:7] | 0]
    const float* __restrict__ zpart,           // [512][2048]
    float* __restrict__ cbuf) {                // [512][512]
  const int lane = threadIdx.x;
  const int m0 = blockIdx.x * 32;
  const int j0 = blockIdx.y * 16;
  v8f ai0 = vzero8(), af0 = vzero8(), ag0 = vzero8(), ao0 = vzero8();
  v8f ai1 = vzero8(), af1 = vzero8(), ag1 = vzero8(), ao1 = vzero8();
  for (int kt = 0; kt < 17; ++kt) {
    const int k0 = kt * 32;
    v16bf A0 = load_tile16x32(xh_r, XHLD, m0,      k0, lane);
    v16bf A1 = load_tile16x32(xh_r, XHLD, m0 + 16, k0, lane);
    v16bf Bi = load_tile16x32(Wc, XHLD, j0,           k0, lane);
    v16bf Bf = load_tile16x32(Wc, XHLD, j0 + HID,     k0, lane);
    v16bf Bg = load_tile16x32(Wc, XHLD, j0 + 2*HID,   k0, lane);
    v16bf Bo = load_tile16x32(Wc, XHLD, j0 + 3*HID,   k0, lane);
    ai0 = wmma_bf16(A0, Bi, ai0);  ai1 = wmma_bf16(A1, Bi, ai1);
    af0 = wmma_bf16(A0, Bf, af0);  af1 = wmma_bf16(A1, Bf, af1);
    ag0 = wmma_bf16(A0, Bg, ag0);  ag1 = wmma_bf16(A1, Bg, ag1);
    ao0 = wmma_bf16(A0, Bo, ao0);  ao1 = wmma_bf16(A1, Bo, ao1);
  }
  const int col = lane & 15, mb = (lane >> 4) * 8;
#pragma unroll
  for (int half = 0; half < 2; ++half) {
    const v8f& ai = half ? ai1 : ai0;
    const v8f& af = half ? af1 : af0;
    const v8f& ag = half ? ag1 : ag0;
    const v8f& ao = half ? ao1 : ao0;
#pragma unroll
    for (int r = 0; r < 8; ++r) {
      const int m = m0 + half * 16 + mb + r, n = j0 + col;
      const float* zp = zpart + (size_t)m * G4 + n;
      float gi = ai[r] + zp[0];
      float gf = af[r] + zp[HID];
      float gg = ag[r] + zp[2 * HID];
      float go = ao[r] + zp[3 * HID];
      float c_old = cbuf[(size_t)m * HID + n];
      float cn = sigm(gf) * c_old + sigm(gi) * tanhf(gg);
      float h  = sigm(go) * tanhf(cn);
      cbuf[(size_t)m * HID + n] = cn;
      xh_w[(size_t)m * XHLD + n] = f2bf(h);
    }
  }
}

// Heads for one 16-row batch tile, 8 waves (256 threads):
// stage1: a1 = relu(h@Wa1.T+ba1), o1 = relu(h@Wo1.T+bo1) -> LDS (bf16),
//         each wave handles 4 column-tile PAIRS (a & o share the A load).
// stage2: fused [a1|o1] @ Wao.T (K=1024, N=16, 10 used), K split over 8 waves.
// Writes joints -> d_out joint region, obj -> d_out combined[...,24:27],
// prev_t (bf16) -> xh cols 512..518.
__global__ void __launch_bounds__(256) head_kernel(
    unsigned short* __restrict__ xh,           // h_t in cols 0..511
    const unsigned short* __restrict__ Wa1b, const float* __restrict__ ba1,
    const unsigned short* __restrict__ Wo1b, const float* __restrict__ bo1,
    const unsigned short* __restrict__ Waob,   // [16][1024]
    const float* __restrict__ ba2, const float* __restrict__ bo2,
    const float* __restrict__ jlo, const float* __restrict__ jhi,
    float* __restrict__ out, int t) {
  __shared__ __align__(16) unsigned short ldsA[16 * 1024];
  __shared__ float red[8][16][16];
  const int lane = threadIdx.x & 31;
  const int w    = threadIdx.x >> 5;           // 8 waves
  const int m0   = blockIdx.x * 16;
  const int col = lane & 15, mb = (lane >> 4) * 8;

  // stage 1: wave w covers n-tiles w*4 .. w*4+3, computing a- and o- parts together
  for (int i = 0; i < 4; ++i) {
    const int n0 = (w * 4 + i) * 16;
    v8f acca = vzero8(), acco = vzero8();
    for (int kt = 0; kt < 16; ++kt) {
      v16bf A  = load_tile16x32(xh,   XHLD, m0, kt * 32, lane);
      v16bf Ba = load_tile16x32(Wa1b, HID,  n0, kt * 32, lane);
      v16bf Bo = load_tile16x32(Wo1b, HID,  n0, kt * 32, lane);
      acca = wmma_bf16(A, Ba, acca);
      acco = wmma_bf16(A, Bo, acco);
    }
#pragma unroll
    for (int r = 0; r < 8; ++r) {
      float va = fmaxf(acca[r] + ba1[n0 + col], 0.f);
      float vo = fmaxf(acco[r] + bo1[n0 + col], 0.f);
      ldsA[(size_t)(mb + r) * 1024 + n0 + col]       = f2bf(va);
      ldsA[(size_t)(mb + r) * 1024 + HID + n0 + col] = f2bf(vo);
    }
  }
  __syncthreads();

  // stage 2: K=1024 split across 8 waves (4 K-tiles each), reduce via LDS
  v8f acc = vzero8();
  for (int kt = 0; kt < 4; ++kt) {
    const int k0 = (w * 4 + kt) * 32;
    v16bf A = load_tile16x32(ldsA, 1024, 0, k0, lane);
    v16bf B = load_tile16x32(Waob, 1024, 0, k0, lane);
    acc = wmma_bf16(A, B, acc);
  }
#pragma unroll
  for (int r = 0; r < 8; ++r) red[w][mb + r][col] = acc[r];
  __syncthreads();

  if (w == 0) {
#pragma unroll
    for (int r = 0; r < 8; ++r) {
      float v = 0.f;
#pragma unroll
      for (int p = 0; p < 8; ++p) v += red[p][mb + r][col];
      const int n = col, m = m0 + mb + r;
      if (n < JDIM) {
        float raw = tanhf(v + ba2[n]);
        float rng = (jhi[n] - jlo[n]) * 0.5f;
        float mn  = (jhi[n] + jlo[n]) * 0.5f;
        out[JOINT_BASE + (size_t)m * (TSTEPS * JDIM) + (size_t)t * JDIM + n] = raw * rng + mn;
        xh[(size_t)m * XHLD + HID + n] = f2bf(raw);   // prev for next step
      } else if (n < JDIM + ODIM) {
        out[(size_t)m * (TSTEPS * COMB) + (size_t)t * COMB + FKOUT + (n - JDIM)] = v + bo2[n - JDIM];
      }
    }
  }
}

// agent = joints @ W_fk.T + b_fk  (K=7 -> plain FMA), into combined[...,:24]
__global__ void fk_kernel(const float* __restrict__ Wfk,
                          const float* __restrict__ bfk,
                          float* __restrict__ out) {
  int row = blockIdx.x * 256 + threadIdx.x;
  if (row >= BATCH * TSTEPS) return;
  float jv[JDIM];
  const float* jp = out + JOINT_BASE + (size_t)row * JDIM;
#pragma unroll
  for (int k = 0; k < JDIM; ++k) jv[k] = jp[k];
  float* op = out + (size_t)row * COMB;
#pragma unroll
  for (int o = 0; o < FKOUT; ++o) {
    float s = bfk[o];
#pragma unroll
    for (int k = 0; k < JDIM; ++k) s += Wfk[o * JDIM + k] * jv[k];
    op[o] = s;
  }
}

// ---------------- host ----------------

extern "C" void kernel_launch(void* const* d_in, const int* in_sizes, int n_in,
                              void* d_out, int out_size, void* d_ws, size_t ws_size,
                              hipStream_t stream) {
  (void)in_sizes; (void)n_in; (void)out_size; (void)ws_size;
  const float* z     = (const float*)d_in[0];
  const float* W_zh  = (const float*)d_in[1];
  const float* b_zh  = (const float*)d_in[2];
  const float* W_zc  = (const float*)d_in[3];
  const float* b_zc  = (const float*)d_in[4];
  const float* W_ih  = (const float*)d_in[5];
  const float* W_hh  = (const float*)d_in[6];
  const float* b_ih  = (const float*)d_in[7];
  const float* b_hh  = (const float*)d_in[8];
  const float* Wo1   = (const float*)d_in[9];
  const float* bo1   = (const float*)d_in[10];
  const float* Wo2   = (const float*)d_in[11];
  const float* bo2   = (const float*)d_in[12];
  const float* Wa1   = (const float*)d_in[13];
  const float* ba1   = (const float*)d_in[14];
  const float* Wa2   = (const float*)d_in[15];
  const float* ba2   = (const float*)d_in[16];
  const float* start = (const float*)d_in[17];
  const float* jlo   = (const float*)d_in[18];
  const float* jhi   = (const float*)d_in[19];
  const float* W_fk  = (const float*)d_in[20];
  const float* b_fk  = (const float*)d_in[21];
  float* out = (float*)d_out;

  char* ws = (char*)d_ws;
  size_t off = 0;
  auto take = [&](size_t bytes) {
    void* p = ws + off;
    off = (off + bytes + 255) & ~(size_t)255;
    return p;
  };
  unsigned short* Wc    = (unsigned short*)take((size_t)G4 * XHLD * 2);
  unsigned short* Wihz  = (unsigned short*)take((size_t)G4 * LATD * 2);
  unsigned short* Wa1b  = (unsigned short*)take((size_t)HID * HID * 2);
  unsigned short* Wo1b  = (unsigned short*)take((size_t)HID * HID * 2);
  unsigned short* Waob  = (unsigned short*)take((size_t)16 * 1024 * 2);
  unsigned short* Wzhb  = (unsigned short*)take((size_t)HID * LATD * 2);
  unsigned short* Wzcb  = (unsigned short*)take((size_t)HID * LATD * 2);
  unsigned short* zb    = (unsigned short*)take((size_t)BATCH * LATD * 2);
  float*          zpart = (float*)take((size_t)BATCH * G4 * 4);
  unsigned short* xh0   = (unsigned short*)take((size_t)BATCH * XHLD * 2);
  unsigned short* xh1   = (unsigned short*)take((size_t)BATCH * XHLD * 2);
  float*          cbuf  = (float*)take((size_t)BATCH * HID * 4);
  unsigned short* xh[2] = { xh0, xh1 };

  auto g1 = [](int n) { return dim3((unsigned)((n + 255) / 256)); };

  // pack weights / activations to bf16
  f2bf_copy_kernel<<<g1(HID * HID), 256, 0, stream>>>(Wa1, Wa1b, HID * HID);
  f2bf_copy_kernel<<<g1(HID * HID), 256, 0, stream>>>(Wo1, Wo1b, HID * HID);
  f2bf_copy_kernel<<<g1(HID * LATD), 256, 0, stream>>>(W_zh, Wzhb, HID * LATD);
  f2bf_copy_kernel<<<g1(HID * LATD), 256, 0, stream>>>(W_zc, Wzcb, HID * LATD);
  f2bf_copy_kernel<<<g1(BATCH * LATD), 256, 0, stream>>>(z, zb, BATCH * LATD);
  build_Wc_kernel  <<<g1(G4 * XHLD), 256, 0, stream>>>(Wc, W_hh, W_ih);
  build_Wihz_kernel<<<g1(G4 * LATD), 256, 0, stream>>>(Wihz, W_ih);
  build_Wao_kernel <<<g1(16 * 1024), 256, 0, stream>>>(Waob, Wa2, Wo2);
  init_prev_kernel <<<g1(BATCH * 32), 256, 0, stream>>>(xh0, xh1, start, jlo, jhi);

  // h0 / c0 and the z-dependent gate prefix (hoisted out of the recurrence)
  init_hc_kernel<<<dim3(BATCH / 16, HID / 16), 32, 0, stream>>>(
      zb, Wzhb, Wzcb, b_zh, b_zc, xh0, cbuf);
  zpart_kernel<<<dim3(BATCH / 16, G4 / 16), 32, 0, stream>>>(
      zb, Wihz, b_ih, b_hh, zpart);

  // sequential decode loop (ping-pong h buffers; stream order serializes steps)
  for (int t = 0; t < TSTEPS; ++t) {
    unsigned short* xr = xh[t & 1];
    unsigned short* xw = xh[(t + 1) & 1];
    lstm_gates_kernel<<<dim3(BATCH / 32, HID / 16), 32, 0, stream>>>(
        xr, xw, Wc, zpart, cbuf);
    head_kernel<<<dim3(BATCH / 16), 256, 0, stream>>>(
        xw, Wa1b, ba1, Wo1b, bo1, Waob, ba2, bo2, jlo, jhi, out, t);
  }

  // FK projection fills combined[..., 0:24]
  fk_kernel<<<g1(BATCH * TSTEPS), 256, 0, stream>>>(W_fk, b_fk, out);
}
